// DSAEncoder_23570780520576
// MI455X (gfx1250) — compile-verified, matrix operand order checked by reading, not run on hardware
//
#include <hip/hip_runtime.h>
#include <cmath>

typedef __attribute__((ext_vector_type(16))) _Float16 v16h;
typedef __attribute__((ext_vector_type(8)))  float    v8f;
typedef __attribute__((ext_vector_type(4)))  int      i4;

#define CDIV(a, b) (((a) + (b) - 1) / (b))

// ---- CDNA5 async global->LDS path (ASYNCcnt), with compile-safe fallback ----
#if defined(__HIP_DEVICE_COMPILE__) && \
    __has_builtin(__builtin_amdgcn_global_load_async_to_lds_b128) && \
    __has_builtin(__builtin_amdgcn_s_wait_asynccnt)
#define USE_ASYNC_LDS 1
#else
#define USE_ASYNC_LDS 0
#endif

#if USE_ASYNC_LDS
typedef __attribute__((address_space(1))) i4 gas_i4;
typedef __attribute__((address_space(3))) i4 las_i4;
__device__ __forceinline__ void async_ld16(const void* g, void* l) {
    __builtin_amdgcn_global_load_async_to_lds_b128(
        (gas_i4*)(__attribute__((address_space(1))) void*)(void*)g,
        (las_i4*)(__attribute__((address_space(3))) void*)l, 0, 0);
}
#endif

__device__ __forceinline__ float gelu_f(float x) {
    return 0.5f * x * (1.0f + erff(x * 0.70710678118654752f));
}

// -------------------- conv 3x3 stride2 pad1 + BN(inference) + GELU --------------------
__global__ void conv3x3_s2_bn_gelu(const float* __restrict__ in, const float* __restrict__ w,
                                   const float* __restrict__ cb, const float* __restrict__ g,
                                   const float* __restrict__ bb, float* __restrict__ out,
                                   int B, int Cin, int Cout, int Hin, int Win) {
    int Hout = Hin >> 1, Wout = Win >> 1;
    int total = B * Cout * Hout * Wout;
    int idx = blockIdx.x * blockDim.x + threadIdx.x;
    if (idx >= total) return;
    int ow = idx % Wout;
    int oh = (idx / Wout) % Hout;
    int co = (idx / (Wout * Hout)) % Cout;
    int b  = idx / (Wout * Hout * Cout);
    float acc = cb[co];
    for (int ci = 0; ci < Cin; ++ci) {
        const float* ip = in + ((size_t)(b * Cin + ci) * Hin) * Win;
        const float* wp = w + ((co * Cin + ci) * 9);
        for (int kh = 0; kh < 3; ++kh) {
            int ih = oh * 2 - 1 + kh;
            if ((unsigned)ih >= (unsigned)Hin) continue;
            for (int kw = 0; kw < 3; ++kw) {
                int iw = ow * 2 - 1 + kw;
                if ((unsigned)iw >= (unsigned)Win) continue;
                acc += ip[(size_t)ih * Win + iw] * wp[kh * 3 + kw];
            }
        }
    }
    float y = acc * (g[co] * rsqrtf(1.0f + 1e-5f)) + bb[co];
    out[idx] = gelu_f(y);
}

// -------------------- layout shuffles --------------------
__global__ void spat2tok(const float* __restrict__ x, float* __restrict__ t, int B, int D, int HW) {
    int total = B * D * HW;
    int idx = blockIdx.x * blockDim.x + threadIdx.x;
    if (idx >= total) return;
    int n = idx % HW;
    int d = (idx / HW) % D;
    int b = idx / (HW * D);
    t[((size_t)b * HW + n) * D + d] = x[idx];
}

__global__ void tok2spat(const float* __restrict__ t, float* __restrict__ x, int B, int D, int HW) {
    int total = B * D * HW;
    int idx = blockIdx.x * blockDim.x + threadIdx.x;
    if (idx >= total) return;
    int n = idx % HW;
    int d = (idx / HW) % D;
    int b = idx / (HW * D);
    x[idx] = t[((size_t)b * HW + n) * D + d];
}

// -------------------- LayerNorm -> f16 padded --------------------
__global__ void ln_f16(const float* __restrict__ t, const float* __restrict__ g,
                       const float* __restrict__ b, _Float16* __restrict__ out,
                       int rows, int D, int Dpad) {
    int row = blockIdx.x * blockDim.x + threadIdx.x;
    if (row >= rows) return;
    const float* x = t + (size_t)row * D;
    float m = 0.f;
    for (int d = 0; d < D; ++d) m += x[d];
    m /= (float)D;
    float v = 0.f;
    for (int d = 0; d < D; ++d) { float dd = x[d] - m; v += dd * dd; }
    v /= (float)D;
    float rs = rsqrtf(v + 1e-5f);
    _Float16* o = out + (size_t)row * Dpad;
    for (int d = 0; d < D; ++d) o[d] = (_Float16)((x[d] - m) * rs * g[d] + b[d]);
    for (int d = D; d < Dpad; ++d) o[d] = (_Float16)0.f;
}

// -------------------- f32 rows x K -> f16 rows x Kpad (zero pad) --------------------
__global__ void pack_f16(const float* __restrict__ in, _Float16* __restrict__ out,
                         int rows, int K, int Kpad) {
    int total = rows * Kpad;
    int idx = blockIdx.x * blockDim.x + threadIdx.x;
    if (idx >= total) return;
    int k = idx % Kpad;
    int r = idx / Kpad;
    out[idx] = (k < K) ? (_Float16)in[(size_t)r * K + k] : (_Float16)0.f;
}

// -------------------- split QKV, fold 1/sqrt(hd) into Q, pad head dim to 32 --------------------
__global__ void split_qkv(const float* __restrict__ qkv, _Float16* __restrict__ q16,
                          _Float16* __restrict__ k16, float* __restrict__ v32,
                          int B, int N, int D, int H, int hd, float qscale) {
    int total = B * H * N * 32;
    int idx = blockIdx.x * blockDim.x + threadIdx.x;
    if (idx >= total) return;
    int d = idx & 31;
    int t2 = idx >> 5;
    int n = t2 % N;
    int h = (t2 / N) % H;
    int b = t2 / (N * H);
    size_t o = ((size_t)(b * H + h) * N + n) * 32 + d;
    float qv = 0.f, kv = 0.f;
    if (d < hd) {
        size_t base = ((size_t)(b * N + n)) * 3 * D + h * hd + d;
        qv = qkv[base] * qscale;
        kv = qkv[base + D];
        v32[((size_t)(b * H + h) * N + n) * hd + d] = qkv[base + 2 * D];
    }
    q16[o] = (_Float16)qv;
    k16[o] = (_Float16)kv;
}

// -------------------- generic 16x16 WMMA GEMM: C = A(MxKpad) @ Bw(NxKpad)^T --------------------
// mode 0: Cf = A@B^T + bias
// mode 1: Cf = resid + A@B^T + bias
// mode 2: Ch = f16(gelu(A@B^T + bias))
__global__ void gemm16(const _Float16* __restrict__ A, const _Float16* __restrict__ Bw,
                       const float* __restrict__ bias, const float* __restrict__ resid,
                       float* __restrict__ Cf, _Float16* __restrict__ Ch,
                       int M, int N, int Kpad, int mode) {
    __shared__ __align__(16) _Float16 lA[16][32];
    __shared__ __align__(16) _Float16 lB[16][32];
    int n0 = blockIdx.x * 16, m0 = blockIdx.y * 16;
    int lane = threadIdx.x;
    int r = lane >> 1, co = (lane & 1) * 16;
    int half = lane >> 4, mn = lane & 15;
    (void)M;
    v8f acc = {0.f, 0.f, 0.f, 0.f, 0.f, 0.f, 0.f, 0.f};
    for (int kt = 0; kt < Kpad; kt += 32) {
        const _Float16* ap = A + (size_t)(m0 + r) * Kpad + kt + co;
        const _Float16* bp = Bw + (size_t)(n0 + r) * Kpad + kt + co;
#if USE_ASYNC_LDS
        // global -> LDS direct (no VGPR roundtrip), tracked by ASYNCcnt
        async_ld16(ap,     &lA[r][co]);
        async_ld16(ap + 8, &lA[r][co + 8]);
        async_ld16(bp,     &lB[r][co]);
        async_ld16(bp + 8, &lB[r][co + 8]);
        __builtin_amdgcn_s_wait_asynccnt(0);
#else
        *(uint4*)&lA[r][co]     = *(const uint4*)ap;
        *(uint4*)&lA[r][co + 8] = *(const uint4*)(ap + 8);
        *(uint4*)&lB[r][co]     = *(const uint4*)bp;
        *(uint4*)&lB[r][co + 8] = *(const uint4*)(bp + 8);
#endif
        __syncthreads();
        union { v16h v; _Float16 e[16]; } ua, ub;
        for (int h2 = 0; h2 < 16; ++h2) {
            int k = h2 + ((h2 >= 8) ? 8 : 0) + (half ? 8 : 0);
            ua.e[h2] = lA[mn][k];
            ub.e[h2] = lB[mn][k];
        }
        acc = __builtin_amdgcn_wmma_f32_16x16x32_f16(false, ua.v, false, ub.v,
                                                     (short)0, acc, false, false);
        __syncthreads();
    }
    union { v8f v; float e[8]; } uc;
    uc.v = acc;
    int col = n0 + mn;
    int rbase = m0 + half * 8;
    float bc = bias[col];
    for (int v = 0; v < 8; ++v) {
        size_t oi = (size_t)(rbase + v) * N + col;
        float val = uc.e[v] + bc;
        if (mode == 1) val += resid[oi];
        if (mode == 2) Ch[oi] = (_Float16)gelu_f(val);
        else           Cf[oi] = val;
    }
}

// -------------------- streaming sparse attention: WMMA scores + in-LDS top-k --------------------
// grid = B*H*(N/16) blocks, 32 threads (1 wave). Never materializes the NxN score tensor.
// K tiles are double-buffered in LDS via async global->LDS loads (ASYNCcnt overlap).
__global__ void dsa_attn(const _Float16* __restrict__ q16, const _Float16* __restrict__ k16,
                         const float* __restrict__ v32, float* __restrict__ obuf,
                         int B, int H, int N, int hd, int kk, int D) {
    __shared__ __align__(16) _Float16 lQ[16][32];
    __shared__ __align__(16) _Float16 lK[2][16][32];
    __shared__ float lS[16][16];
    __shared__ float tvv[16][64];
    __shared__ int   tii[16][64];
    __shared__ float mnv[16];
    __shared__ int   mnp[16];
    int nt = N >> 4;
    int bx = blockIdx.x;
    int qt = bx % nt;
    int h  = (bx / nt) % H;
    int b  = bx / (nt * H);
    size_t bh = (size_t)(b * H + h) * N;
    int lane = threadIdx.x;
    int r = lane >> 1, co = (lane & 1) * 16;
    int half = lane >> 4, mn = lane & 15;

    {   // load Q tile once (16 queries x 32 padded head dims)
        const _Float16* qp = q16 + (bh + qt * 16 + r) * 32 + co;
        *(uint4*)&lQ[r][co]     = *(const uint4*)qp;
        *(uint4*)&lQ[r][co + 8] = *(const uint4*)(qp + 8);
    }
    __syncthreads();
    union U16 { v16h v; _Float16 e[16]; };
    U16 ua;
    for (int h2 = 0; h2 < 16; ++h2) {
        int k = h2 + ((h2 >= 8) ? 8 : 0) + (half ? 8 : 0);
        ua.e[h2] = lQ[mn][k];
    }

#if USE_ASYNC_LDS
    // each lane moves 2 x 16B of its (row, half-row) slice; 2 async ops per tile
    auto issueK = [&](int kt, int buf) {
        const char* gp = (const char*)(k16 + (bh + (size_t)kt * 16 + r) * 32) + (lane & 1) * 16;
        char* lp = (char*)&lK[buf][r][0] + (lane & 1) * 16;
        async_ld16(gp, lp);
        async_ld16(gp + 32, lp + 32);
    };
    issueK(0, 0);
#endif

    for (int kt = 0; kt < nt; ++kt) {
#if USE_ASYNC_LDS
        if (kt + 1 < nt) {
            issueK(kt + 1, (kt + 1) & 1);              // prefetch next tile into other buffer
            __builtin_amdgcn_s_wait_asynccnt(2);       // in-order: current tile's 2 ops landed
        } else {
            __builtin_amdgcn_s_wait_asynccnt(0);
        }
        const _Float16 (*lKc)[32] = lK[kt & 1];
#else
        {
            const _Float16* kp = k16 + (bh + (size_t)kt * 16 + r) * 32 + co;
            *(uint4*)&lK[0][r][co]     = *(const uint4*)kp;
            *(uint4*)&lK[0][r][co + 8] = *(const uint4*)(kp + 8);
            if (kt + 1 < nt) __builtin_prefetch(kp + 16 * 32, 0, 3);
        }
        const _Float16 (*lKc)[32] = lK[0];
#endif
        __syncthreads();
        U16 ub;
        for (int h2 = 0; h2 < 16; ++h2) {
            int k = h2 + ((h2 >= 8) ? 8 : 0) + (half ? 8 : 0);
            ub.e[h2] = lKc[mn][k];
        }
        v8f z = {0.f, 0.f, 0.f, 0.f, 0.f, 0.f, 0.f, 0.f};
        v8f c = __builtin_amdgcn_wmma_f32_16x16x32_f16(false, ua.v, false, ub.v,
                                                       (short)0, z, false, false);
        union { v8f v; float e[8]; } uc;
        uc.v = c;
        for (int v = 0; v < 8; ++v) lS[half * 8 + v][mn] = uc.e[v];
        __syncthreads();
        if (lane < 16) {  // one lane per query row: streaming top-k (replace-min)
            int m = lane;
            for (int j = 0; j < 16; ++j) {
                float s = lS[m][j];
                int cnt = kt * 16 + j;  // items seen before this one == its global key index
                if (cnt < kk) {
                    tvv[m][cnt] = s;
                    tii[m][cnt] = cnt;
                    if (cnt == kk - 1) {
                        float mv = tvv[m][0]; int mp = 0;
                        for (int s2 = 1; s2 < kk; ++s2)
                            if (tvv[m][s2] < mv) { mv = tvv[m][s2]; mp = s2; }
                        mnv[m] = mv; mnp[m] = mp;
                    }
                } else if (s > mnv[m]) {
                    int p = mnp[m];
                    tvv[m][p] = s;
                    tii[m][p] = cnt;
                    float mv = tvv[m][0]; int mp = 0;
                    for (int s2 = 1; s2 < kk; ++s2)
                        if (tvv[m][s2] < mv) { mv = tvv[m][s2]; mp = s2; }
                    mnv[m] = mv; mnp[m] = mp;
                }
            }
        }
        __syncthreads();
    }

    if (lane < 16) {  // softmax over top-k + gather V
        int m = lane;
        int kused = (kk < N) ? kk : N;
        float mx = -3.4e38f;
        for (int s = 0; s < kused; ++s) mx = fmaxf(mx, tvv[m][s]);
        float sum = 0.f;
        for (int s = 0; s < kused; ++s) { float e = expf(tvv[m][s] - mx); tvv[m][s] = e; sum += e; }
        float inv = 1.f / sum;
        float acc[16];
        for (int d = 0; d < 16; ++d) acc[d] = 0.f;
        for (int s = 0; s < kused; ++s) {
            float w = tvv[m][s] * inv;
            const float* vr = v32 + (bh + tii[m][s]) * hd;
            for (int d = 0; d < hd; ++d) acc[d] += w * vr[d];
        }
        int q = qt * 16 + m;
        float* op = obuf + ((size_t)b * N + q) * D + h * hd;
        for (int d = 0; d < hd; ++d) op[d] = acc[d];
    }
}

// -------------------- host orchestration --------------------
struct BlockP {
    const float *ln1_g, *ln1_b, *qkv_w, *qkv_b, *proj_w, *proj_b;
    const float *ln2_g, *ln2_b, *fc1_w, *fc1_b, *fc2_w, *fc2_b;
};

extern "C" void kernel_launch(void* const* d_in, const int* in_sizes, int n_in,
                              void* d_out, int out_size, void* d_ws, size_t ws_size,
                              hipStream_t stream) {
    (void)in_sizes; (void)n_in; (void)out_size; (void)ws_size;
    int ii = 0;
    auto nxt = [&]() { return (const float*)d_in[ii++]; };
    const float* x = nxt();
    const float *c1w = nxt(), *c1b = nxt(), *bn1g = nxt(), *bn1b = nxt();
    const float *c2w = nxt(), *c2b = nxt(), *bn2g = nxt(), *bn2b = nxt();
    auto getBlock = [&]() {
        BlockP p;
        p.ln1_g = nxt(); p.ln1_b = nxt(); p.qkv_w = nxt(); p.qkv_b = nxt();
        p.proj_w = nxt(); p.proj_b = nxt(); p.ln2_g = nxt(); p.ln2_b = nxt();
        p.fc1_w = nxt(); p.fc1_b = nxt(); p.fc2_w = nxt(); p.fc2_b = nxt();
        return p;
    };
    BlockP blk1 = getBlock();
    const float *ds2w = nxt(), *ds2b = nxt(), *ds2g = nxt(), *ds2bb = nxt();
    BlockP blk2 = getBlock();
    const float *ds3w = nxt(), *ds3b = nxt(), *ds3g = nxt(), *ds3bb = nxt();
    BlockP blk3 = getBlock();

    // workspace arena (max-size buffers, reused across stages; ~9.5 MB total)
    char* wsb = (char*)d_ws;
    size_t off = 0;
    auto alloc = [&](size_t bytes) -> void* {
        void* p = wsb + off;
        off += (bytes + 255) & ~(size_t)255;
        return p;
    };
    float*     conv1buf = (float*)alloc((size_t)2 * 8 * 128 * 128 * 4);
    float*     spat     = (float*)alloc((size_t)2 * 16 * 64 * 64 * 4);
    float*     tA       = (float*)alloc((size_t)2 * 4096 * 16 * 4);
    float*     tB       = (float*)alloc((size_t)2 * 4096 * 16 * 4);
    _Float16*  h16      = (_Float16*)alloc((size_t)2 * 4096 * 64 * 2);
    float*     qkvb     = (float*)alloc((size_t)2 * 4096 * 48 * 4);
    _Float16*  q16      = (_Float16*)alloc((size_t)2 * 2 * 4096 * 32 * 2);
    _Float16*  k16      = (_Float16*)alloc((size_t)2 * 2 * 4096 * 32 * 2);
    float*     v32      = (float*)alloc((size_t)2 * 4096 * 16 * 4);
    float*     obuf     = (float*)alloc((size_t)2 * 4096 * 16 * 4);
    _Float16*  mlp16    = (_Float16*)alloc((size_t)2 * 4096 * 64 * 2);
    _Float16*  wq16     = (_Float16*)alloc(32768);
    _Float16*  wp16     = (_Float16*)alloc(32768);
    _Float16*  wf1      = (_Float16*)alloc(32768);
    _Float16*  wf2      = (_Float16*)alloc(32768);

    auto run_block = [&](const BlockP& p, int B, int N, int D, int H, int tk) {
        int BN = B * N, D3 = 3 * D, D4 = 4 * D, hd = D / H;
        int Dpad = (D + 31) / 32 * 32;
        int kk = tk < N ? tk : N;
        float qscale = 1.0f / sqrtf((float)hd);
        ln_f16<<<CDIV(BN, 256), 256, 0, stream>>>(tA, p.ln1_g, p.ln1_b, h16, BN, D, Dpad);
        pack_f16<<<CDIV(D3 * Dpad, 256), 256, 0, stream>>>(p.qkv_w, wq16, D3, D, Dpad);
        gemm16<<<dim3(D3 / 16, BN / 16), 32, 0, stream>>>(h16, wq16, p.qkv_b, nullptr,
                                                          qkvb, nullptr, BN, D3, Dpad, 0);
        split_qkv<<<CDIV(B * H * N * 32, 256), 256, 0, stream>>>(qkvb, q16, k16, v32,
                                                                 B, N, D, H, hd, qscale);
        dsa_attn<<<B * H * (N / 16), 32, 0, stream>>>(q16, k16, v32, obuf, B, H, N, hd, kk, D);
        pack_f16<<<CDIV(BN * Dpad, 256), 256, 0, stream>>>(obuf, h16, BN, D, Dpad);
        pack_f16<<<CDIV(D * Dpad, 256), 256, 0, stream>>>(p.proj_w, wp16, D, D, Dpad);
        gemm16<<<dim3(D / 16, BN / 16), 32, 0, stream>>>(h16, wp16, p.proj_b, tA,
                                                         tB, nullptr, BN, D, Dpad, 1);
        ln_f16<<<CDIV(BN, 256), 256, 0, stream>>>(tB, p.ln2_g, p.ln2_b, h16, BN, D, Dpad);
        pack_f16<<<CDIV(D4 * Dpad, 256), 256, 0, stream>>>(p.fc1_w, wf1, D4, D, Dpad);
        gemm16<<<dim3(D4 / 16, BN / 16), 32, 0, stream>>>(h16, wf1, p.fc1_b, nullptr,
                                                          nullptr, mlp16, BN, D4, Dpad, 2);
        pack_f16<<<CDIV(D * D4, 256), 256, 0, stream>>>(p.fc2_w, wf2, D, D4, D4);
        gemm16<<<dim3(D / 16, BN / 16), 32, 0, stream>>>(mlp16, wf2, p.fc2_b, tB,
                                                         tA, nullptr, BN, D, D4, 1);
    };

    float* out = (float*)d_out;

    // patch embed
    conv3x3_s2_bn_gelu<<<CDIV(2 * 8 * 128 * 128, 256), 256, 0, stream>>>(
        x, c1w, c1b, bn1g, bn1b, conv1buf, 2, 1, 8, 256, 256);
    conv3x3_s2_bn_gelu<<<CDIV(2 * 16 * 64 * 64, 256), 256, 0, stream>>>(
        conv1buf, c2w, c2b, bn2g, bn2b, spat, 2, 8, 16, 128, 128);

    // stage 1 (N=4096, D=16, H=2, topk=64) -> f1
    spat2tok<<<CDIV(2 * 16 * 4096, 256), 256, 0, stream>>>(spat, tA, 2, 16, 4096);
    run_block(blk1, 2, 4096, 16, 2, 64);
    tok2spat<<<CDIV(2 * 16 * 4096, 256), 256, 0, stream>>>(tA, out, 2, 16, 4096);

    // stage 2 (downsample, N=1024, D=32, H=2, topk=32) -> f2
    conv3x3_s2_bn_gelu<<<CDIV(2 * 32 * 32 * 32, 256), 256, 0, stream>>>(
        out, ds2w, ds2b, ds2g, ds2bb, spat, 2, 16, 32, 64, 64);
    spat2tok<<<CDIV(2 * 32 * 1024, 256), 256, 0, stream>>>(spat, tA, 2, 32, 1024);
    run_block(blk2, 2, 1024, 32, 2, 32);
    tok2spat<<<CDIV(2 * 32 * 1024, 256), 256, 0, stream>>>(tA, out + 131072, 2, 32, 1024);

    // stage 3 (downsample, N=256, D=48, H=4, topk=16) -> f3
    conv3x3_s2_bn_gelu<<<CDIV(2 * 48 * 16 * 16, 256), 256, 0, stream>>>(
        out + 131072, ds3w, ds3b, ds3g, ds3bb, spat, 2, 32, 48, 32, 32);
    spat2tok<<<CDIV(2 * 48 * 256, 256), 256, 0, stream>>>(spat, tA, 2, 48, 256);
    run_block(blk3, 2, 256, 48, 4, 16);
    tok2spat<<<CDIV(2 * 48 * 256, 256), 256, 0, stream>>>(tA, out + 131072 + 65536, 2, 48, 256);
}
